// FLASHTransformer_22660247454341
// MI455X (gfx1250) — matrix operand order repaired and checked by reading, not compiled, not run on hardware
//
#include <hip/hip_runtime.h>
#include <hip/hip_bf16.h>
#include <stdint.h>

// ---------------- problem constants ----------------
#define BB   512
#define NIN  128
#define DIM  512
#define QKD  128
#define GSZ  32
#define LYR  2
#define BINS 100
#define HID  1024
#define NG   4          // NIN / GSZ
#define ROT  32
#define TT   (BB * NIN) // 65536 tokens
#define EPSF 1e-5f

typedef __attribute__((ext_vector_type(16))) __bf16   v16bf;
typedef __attribute__((ext_vector_type(8)))  float    v8f;
typedef __attribute__((ext_vector_type(4)))  uint32_t v4u;
typedef __attribute__((ext_vector_type(8)))  int      v8i;
typedef __attribute__((ext_vector_type(4)))  int      v4i;

// ---------------- bf16 helpers ----------------
__device__ __forceinline__ __bf16 f2bf(float f) {
    uint32_t u = __builtin_bit_cast(uint32_t, f);
    uint32_t r = u + 0x7FFFu + ((u >> 16) & 1u);   // round-to-nearest-even
    unsigned short h = (unsigned short)(r >> 16);
    return __builtin_bit_cast(__bf16, h);
}
__device__ __forceinline__ float bf2f(__bf16 b) {
    unsigned short h = __builtin_bit_cast(unsigned short, b);
    uint32_t u = ((uint32_t)h) << 16;
    return __builtin_bit_cast(float, u);
}
// fast silu: x * v_rcp_f32(1+exp(-x))  (avoids IEEE div expansion)
__device__ __forceinline__ float silu(float x) {
    return x * __builtin_amdgcn_rcpf(1.f + __expf(-x));
}

// ---------------- WMMA fragment layout (16-bit, contiguous k order used by this file) ----------------
// A 16x32: lane half h (l>>4), m = l&15. Element e ->
//   k = e + 16h   (one contiguous 16-run per lane, matches B fragments)
__device__ __forceinline__ int akidx(int e, int half) {
    return e + 16 * half;
}
// A[m][k], row-major storage, leading dim ld (per-lane: one contiguous 32B run)
__device__ __forceinline__ v16bf fragA(const __bf16* p, int ld, int m0, int k0) {
    int lane = threadIdx.x & 31, half = lane >> 4, m = m0 + (lane & 15);
    v16bf r;
#pragma unroll
    for (int e = 0; e < 16; ++e) r[e] = p[(size_t)m * ld + k0 + akidx(e, half)];
    return r;
}
// B 32x16: lane n = l&15, k = e + 16*(l>>4).
// Storage TRANSPOSED: element (k,n) at p[n*ld + k]  (per-lane contiguous 32B run)
__device__ __forceinline__ v16bf fragBT(const __bf16* p, size_t ld, int k0, int n0) {
    int lane = threadIdx.x & 31, half = lane >> 4, n = n0 + (lane & 15);
    v16bf r;
#pragma unroll
    for (int e = 0; e < 16; ++e) r[e] = p[(size_t)n * ld + k0 + e + 16 * half];
    return r;
}
// B fragment from row-major f32 storage (lin_kv state in LDS), convert on read
__device__ __forceinline__ v16bf fragBf32(const float* p, int ld, int k0, int n0) {
    int lane = threadIdx.x & 31, half = lane >> 4, n = n0 + (lane & 15);
    v16bf r;
#pragma unroll
    for (int e = 0; e < 16; ++e) r[e] = f2bf(p[(k0 + e + 16 * half) * ld + n]);
    return r;
}
__device__ __forceinline__ v8f wmma_bf(v16bf a, v16bf b, v8f c) {
    return __builtin_amdgcn_wmma_f32_16x16x32_bf16(false, a, false, b, (short)0, c,
                                                   false, false);
}

// ---------------- CDNA5 async / TDM primitives ----------------
// async global->LDS, 16B per lane (ASYNCcnt)
__device__ __forceinline__ void async_ld_b128(uint32_t lds_off, const void* gaddr) {
    asm volatile("global_load_async_to_lds_b128 %0, %1, off"
                 :: "v"(lds_off), "v"(gaddr) : "memory");
}
__device__ __forceinline__ void wait_asynccnt0() {
    asm volatile("s_wait_asynccnt 0x0" ::: "memory");
}
// TDM: 2D tile (lines x elems, bf16) from global [lines total, stride elems] -> LDS
__device__ __forceinline__ void tdm_load_2d_bf16(uint32_t lds_off, const void* gtile,
                                                 uint32_t elems, uint32_t lines,
                                                 uint32_t stride_elems,
                                                 uint32_t tensor_d0, uint32_t tensor_d1) {
    uint64_t ga = (uint64_t)(uintptr_t)gtile;
    v4u g0;
    g0[0] = 1u;                                           // count=1, user mode
    g0[1] = lds_off;                                      // lds_addr
    g0[2] = (uint32_t)(ga & 0xFFFFFFFFu);                 // global_addr[31:0]
    g0[3] = (uint32_t)((ga >> 32) & 0x1FFFFFFu) | (2u << 30); // addr[56:32] | type=2
    v8i g1;
    g1[0] = (int)(1u << 16);                              // wg_mask=0, data_size=1 (2B)
    g1[1] = (int)((tensor_d0 & 0xFFFFu) << 16);           // tensor_dim0[15:0]
    g1[2] = (int)(((tensor_d0 >> 16) & 0xFFFFu) | ((tensor_d1 & 0xFFFFu) << 16));
    g1[3] = (int)(((tensor_d1 >> 16) & 0xFFFFu) | ((elems & 0xFFFFu) << 16)); // tile_dim0
    g1[4] = (int)(lines & 0xFFFFu);                       // tile_dim1 (tile_dim2=0)
    g1[5] = (int)stride_elems;                            // tensor_dim0_stride[31:0]
    g1[6] = 0;                                            // stride[47:32], dim1_stride lo
    g1[7] = 0;
    v4i g2 = {}; v4i g3 = {};
#if __has_builtin(__builtin_amdgcn_tensor_load_to_lds)
#if defined(__clang_major__) && (__clang_major__ >= 23)
    v8i z8 = {};
    __builtin_amdgcn_tensor_load_to_lds(g0, g1, g2, g3, z8, 0);
#else
    __builtin_amdgcn_tensor_load_to_lds(g0, g1, g2, g3, 0);
#endif
#else
    asm volatile("tensor_load_to_lds %0, %1, %2, %3"
                 :: "s"(g0), "s"(g1), "s"(g2), "s"(g3) : "memory");
#endif
}
__device__ __forceinline__ void wait_tensorcnt0() {
    __builtin_amdgcn_s_wait_tensorcnt((short)0);
}

// ---------------- small elementwise kernels ----------------
// weight convert f32 [K][N] -> bf16 TRANSPOSED [N][K]
__global__ void k_wcvt(const float* __restrict__ s, __bf16* __restrict__ d,
                       int K, int N) {
    int i = blockIdx.x * 256 + threadIdx.x;    // over N*K, d-linear
    if (i >= N * K) return;
    int n = i / K, k = i - n * K;
    d[i] = f2bf(s[(size_t)k * N + n]);
}

__global__ void k_embed(const int* __restrict__ tok, const float* __restrict__ emb,
                        const float* __restrict__ pos, float* __restrict__ x) {
    int i = blockIdx.x * 256 + threadIdx.x;    // over TT*DIM
    int d = i & (DIM - 1);
    int t = i >> 9;                             // token index b*NIN+n
    int n = t & (NIN - 1);
    int tk = tok[t];
    x[i] = emb[((size_t)n * BINS + tk) * DIM + d] + pos[n * DIM + d];
}

// RMS-ish norm: one wave per token (wave32 shuffle reduce)
__global__ void k_norm(const float* __restrict__ x, __bf16* __restrict__ nx,
                       const float* __restrict__ ngp, int l) {
    int wv = threadIdx.x >> 5, lane = threadIdx.x & 31;
    size_t t = (size_t)blockIdx.x * 8 + wv;
    const float* row = x + t * DIM;
    float ss = 0.f;
    for (int i = lane; i < DIM; i += 32) { float v = row[i]; ss += v * v; }
#pragma unroll
    for (int m = 16; m; m >>= 1) ss += __shfl_xor(ss, m, 32);
    float nrm = sqrtf(ss) * 0.044194173824159216f;   // * DIM^-0.5
    float sc  = ngp[l] / fmaxf(nrm, EPSF);
    __bf16* o = nx + t * DIM;
    for (int i = lane; i < DIM; i += 32) o[i] = f2bf(row[i] * sc);
}

// token-shift: first DIM/2 channels come from previous token (0 at n==0)
__global__ void k_shift(const __bf16* __restrict__ nx, __bf16* __restrict__ o) {
    int i = blockIdx.x * 256 + threadIdx.x;    // over TT*DIM
    int d = i & (DIM - 1);
    int t = i >> 9;
    int n = t & (NIN - 1);
    if (d < DIM / 2) o[i] = (n == 0) ? f2bf(0.f) : nx[i - DIM];
    else             o[i] = nx[i];
}

// ---------------- tiled bf16 WMMA GEMM ----------------
// A [M,K] row-major bf16 (activations), W [N,K] bf16 (pre-transposed weights).
// Block tile 64(M) x 128(N), K-step 32, 8 waves (2x4), wave tile 32x32.
// A tile -> LDS via global_load_async_to_lds_b128 (ASYNCcnt).
// W tile (128 lines x 32 elems, stride K) -> LDS via Tensor Data Mover (TENSORcnt).
// mode 0: silu(acc+bias) -> outV/outG TRANSPOSED [col][row]   (Wh GEMM)
// mode 1: silu(acc+bias) -> outF f32 row-major                 (Wqk GEMM)
// mode 2: acc+bias + outF residual -> outF f32 row-major       (Wout GEMM)
__global__ __launch_bounds__(256) void k_gemm(const __bf16* __restrict__ A,
                                              const __bf16* __restrict__ Wt,
                                              const float* __restrict__ bias,
                                              int K, int N, int mode,
                                              float* outF, __bf16* outV, __bf16* outG) {
    __shared__ __bf16 As[64 * 32];       // [m][k]
    __shared__ __bf16 Bs[128 * 32];      // [n][k]  (transposed tile, TDM-filled)
    int bn = blockIdx.x, bm = blockIdx.y;
    int tid = threadIdx.x, lane = tid & 31, w = tid >> 5;
    int wm = w >> 2, wn = w & 3;
    v8f acc[2][2] = {};

    for (int kt = 0; kt < K; kt += 32) {
        __syncthreads();                 // previous iteration's reads done
        {   // A tile: 64x32, one async b128 per thread
            int r = tid >> 2, c = (tid & 3) * 8;
            const __bf16* src = A + (size_t)((size_t)bm * 64 + r) * K + kt + c;
            async_ld_b128((uint32_t)(uintptr_t)&As[r * 32 + c], src);
        }
        if (w == 0) {                    // W tile via TDM: 128 lines x 32 elems
            const __bf16* gt = Wt + (size_t)bn * 128 * K + kt;
            tdm_load_2d_bf16((uint32_t)(uintptr_t)Bs, gt, 32, 128, (uint32_t)K,
                             (uint32_t)K, (uint32_t)N);
        }
        wait_asynccnt0();
        if (w == 0) wait_tensorcnt0();
        __syncthreads();
        v16bf a0 = fragA(As, 32, wm * 32 + 0, 0);
        v16bf a1 = fragA(As, 32, wm * 32 + 16, 0);
        v16bf b0 = fragBT(Bs, 32, 0, wn * 32 + 0);
        v16bf b1 = fragBT(Bs, 32, 0, wn * 32 + 16);
        acc[0][0] = wmma_bf(a0, b0, acc[0][0]);
        acc[0][1] = wmma_bf(a0, b1, acc[0][1]);
        acc[1][0] = wmma_bf(a1, b0, acc[1][0]);
        acc[1][1] = wmma_bf(a1, b1, acc[1][1]);
    }

    int half = lane >> 4, nl = lane & 15;
#pragma unroll
    for (int mt = 0; mt < 2; ++mt)
#pragma unroll
        for (int nt = 0; nt < 2; ++nt) {
            int col = bn * 128 + wn * 32 + nt * 16 + nl;
            float bcol = bias[col];
#pragma unroll
            for (int r = 0; r < 8; ++r) {
                size_t row = (size_t)bm * 64 + wm * 32 + mt * 16 + r + 8 * half;
                float v = acc[mt][nt][r] + bcol;
                if (mode == 0) {         // transposed [col][row] stores (16B/lane runs)
                    float s = silu(v);
                    if (col < HID) outV[(size_t)col * TT + row] = f2bf(s);
                    else           outG[(size_t)(col - HID) * TT + row] = f2bf(s);
                } else if (mode == 1) {
                    outF[row * (size_t)N + col] = silu(v);
                } else {
                    size_t idx = row * (size_t)N + col;
                    outF[idx] = v + outF[idx];           // + residual h
                }
            }
        }
}

// ---------------- q/k projection + rotary ----------------
// qq, lq, kk stored [token][d]; lk stored TRANSPOSED [d][token] for lk^T A-frags
__global__ void k_qrot(const float* __restrict__ qk, const float* __restrict__ gm,
                       const float* __restrict__ bt, __bf16* __restrict__ qq,
                       __bf16* __restrict__ lq, __bf16* __restrict__ kk,
                       __bf16* __restrict__ lkT) {
    int i = blockIdx.x * 256 + threadIdx.x;    // over TT*QKD
    int d = i & (QKD - 1);
    size_t t = (size_t)(i >> 7);
    int n = (int)(t & (NIN - 1));
    float x  = qk[i];
    float xp = 0.f, c = 1.f, s = 0.f;
    bool even = !(d & 1);
    if (d < ROT) {
        xp = qk[(t << 7) + (size_t)(d ^ 1)];
        int j = d >> 1;
        float inv = __powf(10000.f, -(2.f * j) / (float)ROT);
        float ang = (float)n * inv;
        c = __cosf(ang); s = __sinf(ang);
    }
#pragma unroll
    for (int q4 = 0; q4 < 4; ++q4) {
        float v = x * gm[q4 * QKD + d] + bt[q4 * QKD + d];
        if (d < ROT) {
            float vp = xp * gm[q4 * QKD + (d ^ 1)] + bt[q4 * QKD + (d ^ 1)];
            v = even ? (v * c - vp * s) : (v * c + vp * s);
        }
        __bf16 bv = f2bf(v);
        if      (q4 == 0) qq[i] = bv;
        else if (q4 == 1) lq[i] = bv;
        else if (q4 == 2) kk[i] = bv;
        else              lkT[(size_t)d * TT + t] = bv;
    }
}

// ---------------- fused grouped attention (quad + linear, exclusive cumsum) ----
__device__ __forceinline__ int t5_bucket(int i, int j) {
    int neg = i - j; if (neg < 0) neg = 0;
    if (neg < 16) return neg;
    float vl = 16.f + (__logf((float)neg * (1.f / 16.f)) * (1.f / __logf(8.f))) * 16.f;
    int b = (int)vl;                               // truncation matches astype(int64)
    return b > 31 ? 31 : b;
}

// grid: (HID/64, B). 64 threads = 2 waves; wave w owns state cols [32w,32w+32).
// v/gate are TRANSPOSED [hid][token]; lk is TRANSPOSED [d][token].
// No barriers needed: all LDS slices are wave-private, LDS in-order per wave.
__global__ __launch_bounds__(64) void k_attn(const __bf16* __restrict__ qq,
                                             const __bf16* __restrict__ lq,
                                             const __bf16* __restrict__ kk,
                                             const __bf16* __restrict__ lkT,
                                             const __bf16* __restrict__ vT,
                                             const __bf16* __restrict__ gT,
                                             const float* __restrict__ rp,
                                             __bf16* __restrict__ a2) {
    __shared__ float  st[128 * 64];      // lin_kv state: QKD x 64-chunk, f32
    __shared__ __bf16 at[2][32 * 32];    // per-wave attn matrix
    int lane = threadIdx.x & 31, w = threadIdx.x >> 5;
    int half = lane >> 4, nl = lane & 15;
    int b  = blockIdx.y;
    int c0 = blockIdx.x * 64;            // hid chunk base
    int wn0 = w * 32;                    // wave's col offset within chunk

    for (int r = lane; r < 128; r += 32)
#pragma unroll
        for (int cc = 0; cc < 32; ++cc) st[r * 64 + wn0 + cc] = 0.f;

    const float t5s = sqrtf((float)QKD);

    for (int g = 0; g < NG; ++g) {
        size_t base = ((size_t)b * NIN + (size_t)g * GSZ);
        const __bf16* qqB = qq + base * QKD;
        const __bf16* lqB = lq + base * QKD;
        const __bf16* kkB = kk + base * QKD;
        const __bf16* lkB = lkT + base;                  // [d][token], ld=TT
        const __bf16* vBt = vT + (size_t)c0 * TT + base; // [n][k], ld=TT
        const __bf16* gBt = gT + (size_t)c0 * TT + base;
        __bf16*       oB  = a2 + base * HID + c0;

        // v chunk B fragments (shared by quad_out and state update)
        v16bf bv0 = fragBT(vBt, TT, 0, wn0 + 0);
        v16bf bv1 = fragBT(vBt, TT, 0, wn0 + 16);

        // sim = qq @ kk^T  (32x32, K=128)
        v8f sim[2][2] = {};
#pragma unroll
        for (int kt = 0; kt < QKD; kt += 32) {
            v16bf a0 = fragA(qqB, QKD, 0, kt);
            v16bf a1 = fragA(qqB, QKD, 16, kt);
            v16bf b0 = fragBT(kkB, QKD, kt, 0);
            v16bf b1 = fragBT(kkB, QKD, kt, 16);
            sim[0][0] = wmma_bf(a0, b0, sim[0][0]);
            sim[0][1] = wmma_bf(a0, b1, sim[0][1]);
            sim[1][0] = wmma_bf(a1, b0, sim[1][0]);
            sim[1][1] = wmma_bf(a1, b1, sim[1][1]);
        }
        // attn = relu(sim/GS + rp[bucket]*sqrt(QK))^2, causal masked
#pragma unroll
        for (int mt = 0; mt < 2; ++mt)
#pragma unroll
            for (int nt = 0; nt < 2; ++nt)
#pragma unroll
                for (int r = 0; r < 8; ++r) {
                    int i = mt * 16 + r + 8 * half, j = nt * 16 + nl;
                    float sv = sim[mt][nt][r] * (1.f / GSZ) + rp[t5_bucket(i, j)] * t5s;
                    sv = fmaxf(sv, 0.f); sv = sv * sv;
                    if (j > i) sv = 0.f;
                    at[w][i * 32 + j] = f2bf(sv);
                }

        // co = attn @ v_chunk (K=32)  +  lq @ lin_kv_state (K=128, pre-update)
        v8f co[2][2] = {};
        {
            v16bf a0 = fragA(at[w], 32, 0, 0);
            v16bf a1 = fragA(at[w], 32, 16, 0);
            co[0][0] = wmma_bf(a0, bv0, co[0][0]);
            co[0][1] = wmma_bf(a0, bv1, co[0][1]);
            co[1][0] = wmma_bf(a1, bv0, co[1][0]);
            co[1][1] = wmma_bf(a1, bv1, co[1][1]);
        }
#pragma unroll
        for (int kt = 0; kt < QKD; kt += 32) {
            v16bf a0 = fragA(lqB, QKD, 0, kt);
            v16bf a1 = fragA(lqB, QKD, 16, kt);
            v16bf b0 = fragBf32(st, 64, kt, wn0 + 0);
            v16bf b1 = fragBf32(st, 64, kt, wn0 + 16);
            co[0][0] = wmma_bf(a0, b0, co[0][0]);
            co[0][1] = wmma_bf(a0, b1, co[0][1]);
            co[1][0] = wmma_bf(a1, b0, co[1][0]);
            co[1][1] = wmma_bf(a1, b1, co[1][1]);
        }

        // state += lk^T @ v_chunk / GS   (128 x 32-slice, K=32)
#pragma unroll
        for (int mt = 0; mt < 8; ++mt) {
            v16bf aT = fragA(lkB, TT, mt * 16, 0);       // lk^T from [d][token]
            v8f d0 = {}, d1 = {};
            d0 = wmma_bf(aT, bv0, d0);
            d1 = wmma_bf(aT, bv1, d1);
#pragma unroll
            for (int r = 0; r < 8; ++r) {
                int drow = mt * 16 + r + 8 * half;
                st[drow * 64 + wn0 + nl]      += d0[r] * (1.f / GSZ);
                st[drow * 64 + wn0 + 16 + nl] += d1[r] * (1.f / GSZ);
            }
        }

        // out = gate * (quad_out + lin_out)
#pragma unroll
        for (int mt = 0; mt < 2; ++mt)
#pragma unroll
            for (int nt = 0; nt < 2; ++nt)
#pragma unroll
                for (int r = 0; r < 8; ++r) {
                    int i  = mt * 16 + r + 8 * half;
                    int cc = wn0 + nt * 16 + nl;
                    float gv = bf2f(gBt[(size_t)cc * TT + i]);
                    oB[(size_t)i * HID + cc] = f2bf(gv * co[mt][nt][r]);
                }
    }
}

// ---------------- final LayerNorm ----------------
__global__ void k_ln(const float* __restrict__ x, const float* __restrict__ g,
                     const float* __restrict__ be, float* __restrict__ out) {
    int wv = threadIdx.x >> 5, lane = threadIdx.x & 31;
    size_t t = (size_t)blockIdx.x * 8 + wv;
    const float* row = x + t * DIM;
    float s = 0.f, ss = 0.f;
    for (int i = lane; i < DIM; i += 32) { float v = row[i]; s += v; ss += v * v; }
#pragma unroll
    for (int m = 16; m; m >>= 1) { s += __shfl_xor(s, m, 32); ss += __shfl_xor(ss, m, 32); }
    float mu  = s * (1.f / DIM);
    float var = ss * (1.f / DIM) - mu * mu;
    float inv = rsqrtf(var + EPSF);
    float* o = out + t * DIM;
    for (int i = lane; i < DIM; i += 32) o[i] = (row[i] - mu) * inv * g[i] + be[i];
}

// ---------------- host orchestration ----------------
extern "C" void kernel_launch(void* const* d_in, const int* in_sizes, int n_in,
                              void* d_out, int out_size, void* d_ws, size_t ws_size,
                              hipStream_t stream) {
    (void)in_sizes; (void)n_in; (void)out_size; (void)ws_size;
    const int*   tokens = (const int*)d_in[0];
    const float* emb    = (const float*)d_in[1];
    const float* pos    = (const float*)d_in[2];
    const float* norm_g = (const float*)d_in[3];
    const float* Wh     = (const float*)d_in[4];
    const float* bh     = (const float*)d_in[5];
    const float* Wqk    = (const float*)d_in[6];
    const float* bqk    = (const float*)d_in[7];
    const float* gamma  = (const float*)d_in[8];
    const float* beta   = (const float*)d_in[9];
    const float* rp     = (const float*)d_in[10];
    const float* Wout   = (const float*)d_in[11];
    const float* bout   = (const float*)d_in[12];
    const float* ln_g   = (const float*)d_in[13];
    const float* ln_b   = (const float*)d_in[14];

    // ---- workspace layout (regions with disjoint lifetimes are aliased) ----
    char* wsp = (char*)d_ws;
    size_t off = 0;
    auto alloc = [&](size_t bytes) -> char* {
        char* p = wsp + off;
        off += (bytes + 255) & ~(size_t)255;
        return p;
    };
    float* x = (float*)alloc((size_t)TT * DIM * 4);              // residual stream
    // union region: {nx, nxs, qk} live stages 1-3; a2 lives stages 4-5
    char* regU = alloc((size_t)2 * TT * DIM * 2 + (size_t)TT * QKD * 4);
    __bf16* nx    = (__bf16*)regU;
    __bf16* nxs   = (__bf16*)(regU + (size_t)TT * DIM * 2);
    float*  qkbuf = (float*)(regU + (size_t)2 * TT * DIM * 2);
    __bf16* a2    = (__bf16*)regU;                               // TT*HID*2 <= region
    __bf16* qqb = (__bf16*)alloc((size_t)TT * QKD * 2);
    __bf16* lqb = (__bf16*)alloc((size_t)TT * QKD * 2);
    __bf16* kkb = (__bf16*)alloc((size_t)TT * QKD * 2);
    __bf16* lkb = (__bf16*)alloc((size_t)TT * QKD * 2);          // transposed [d][t]
    __bf16* vb  = (__bf16*)alloc((size_t)TT * HID * 2);          // transposed [hid][t]
    __bf16* gb  = (__bf16*)alloc((size_t)TT * HID * 2);          // transposed [hid][t]
    __bf16* WhT  = (__bf16*)alloc((size_t)LYR * DIM * 2 * HID * 2);
    __bf16* WqkT = (__bf16*)alloc((size_t)LYR * DIM * QKD * 2);
    __bf16* WoT  = (__bf16*)alloc((size_t)LYR * HID * DIM * 2);

    // weights -> bf16, transposed to [N][K] (every call; deterministic)
    for (int l = 0; l < LYR; ++l) {
        int nWh = DIM * 2 * HID, nWq = DIM * QKD, nWo = HID * DIM;
        k_wcvt<<<(nWh + 255) / 256, 256, 0, stream>>>(
            Wh + (size_t)l * nWh, WhT + (size_t)l * nWh, DIM, 2 * HID);
        k_wcvt<<<(nWq + 255) / 256, 256, 0, stream>>>(
            Wqk + (size_t)l * nWq, WqkT + (size_t)l * nWq, DIM, QKD);
        k_wcvt<<<(nWo + 255) / 256, 256, 0, stream>>>(
            Wout + (size_t)l * nWo, WoT + (size_t)l * nWo, HID, DIM);
    }

    k_embed<<<(TT * DIM) / 256, 256, 0, stream>>>(tokens, emb, pos, x);

    for (int l = 0; l < LYR; ++l) {
        k_norm<<<TT / 8, 256, 0, stream>>>(x, nx, norm_g, l);
        k_shift<<<(TT * DIM) / 256, 256, 0, stream>>>(nx, nxs);
        // hid GEMM: [TT,512] @ [512,2048] -> silu -> vT | gT
        k_gemm<<<dim3((2 * HID) / 128, TT / 64), 256, 0, stream>>>(
            nxs, WhT + (size_t)l * DIM * 2 * HID, bh + (size_t)l * 2 * HID,
            DIM, 2 * HID, 0, nullptr, vb, gb);
        // qk GEMM: [TT,512] @ [512,128] -> silu -> qk f32
        k_gemm<<<dim3(QKD / 128, TT / 64), 256, 0, stream>>>(
            nxs, WqkT + (size_t)l * DIM * QKD, bqk + (size_t)l * QKD,
            DIM, QKD, 1, qkbuf, nullptr, nullptr);
        k_qrot<<<(TT * QKD) / 256, 256, 0, stream>>>(
            qkbuf, gamma + (size_t)l * 4 * QKD, beta + (size_t)l * 4 * QKD,
            qqb, lqb, kkb, lkb);
        k_attn<<<dim3(HID / 64, BB), 64, 0, stream>>>(
            qqb, lqb, kkb, lkb, vb, gb, rp + (size_t)l * 32, a2);
        // out GEMM: [TT,1024] @ [1024,512] + bout + residual -> x
        k_gemm<<<dim3(DIM / 128, TT / 64), 256, 0, stream>>>(
            a2, WoT + (size_t)l * HID * DIM, bout + (size_t)l * DIM,
            HID, DIM, 2, x, nullptr, nullptr);
    }

    k_ln<<<TT / 8, 256, 0, stream>>>(x, ln_g, ln_b, (float*)d_out);
}
